// CrossAttention_52072183497346
// MI455X (gfx1250) — matrix operand order repaired and checked by reading, not compiled, not run on hardware
//
#include <hip/hip_runtime.h>

// ---------------------------------------------------------------------------
// CrossAttention for MI455X (gfx1250, wave32, WMMA).
//
// Roofline: ~120 GFLOP vs ~100MB HBM traffic -> strongly compute bound; all
// four GEMMs run on v_wmma_f32_16x16x32_bf16 (bf16 inputs, f32 accumulate).
//
// Mask note: reference adds NEG_BIG*(1-mask[b,q]) broadcast over the KEY axis
// -> uniform shift per softmax row -> softmax is shift-invariant -> no effect
// on the output. Skipped.
//
// Pipeline (bf16 activations, f32 accum), all staging software-pipelined
// (register prefetch + double-buffered LDS, one barrier per k-step so global
// load latency hides behind the WMMA pipe):
//   proj_gemm x3 : Q = x@Wq, K = ctx@Wk (row-major [tok][d] bf16)
//                  Vt = (ctx@Wv)^T      ([b][d][key] bf16)
//   attn         : per 16-query tile: (1) stats pass (online softmax row
//                  max/sum, wave butterflies + LDS combine),
//                  (2) recompute S, P = exp(s-m)/l, O^T = Vt @ P^T
//                  (Vt fragments stream straight from L2: K/V per batch are
//                  4MB bf16 << 192MB L2).
// Workspace: Q 16MB + K 16MB + Vt 16MB = 48MB bf16.
// ---------------------------------------------------------------------------

typedef __attribute__((ext_vector_type(16))) __bf16 v16bf;
typedef __attribute__((ext_vector_type(8)))  float  v8f;
typedef __attribute__((ext_vector_type(4)))  unsigned int u32x4;
typedef __attribute__((ext_vector_type(4)))  float  f32x4;
typedef unsigned short u16;

#define S_LEN  2048
#define D_DIM  1024
#define NB     4

__device__ __forceinline__ u16 f2bf(float f) {           // RNE fp32->bf16
  unsigned int u = __float_as_uint(f);
  u += 0x7FFFu + ((u >> 16) & 1u);
  return (u16)(u >> 16);
}
__device__ __forceinline__ unsigned int pk2(float a, float b) {
  return (unsigned int)f2bf(a) | ((unsigned int)f2bf(b) << 16);
}

union ABFrag { v16bf v; u32x4 u[2]; };

__device__ __forceinline__ v8f wmma_bf16(const ABFrag& a, const ABFrag& b, v8f c) {
  return __builtin_amdgcn_wmma_f32_16x16x32_bf16(
      false, a.v, false, b.v, (short)0, c, false, false);
}

// ---------------------------------------------------------------------------
// proj_gemm: out[M=8192,N=1024](bf16) = A[M,1024](f32) @ W[1024,1024](f32)+bias
// 256 thr (8 waves). Tile 128x128, k-step 32, waves 4x2 -> 32x64 each.
// Software pipeline: regs hold tile kk+1 while LDS buf[kk&1] feeds WMMAs.
// transpose_v: 0 -> out[tok][n];  1 -> out[b][n][s]  (V stored transposed).
// ---------------------------------------------------------------------------
__global__ __launch_bounds__(256)
void proj_gemm_kernel(const float* __restrict__ A, const float* __restrict__ W,
                      const float* __restrict__ bias, u16* __restrict__ out,
                      int transpose_v) {
  __shared__ __align__(16) u16 ldsA[2][128 * 32];   // [buf][row][k]  16KB
  __shared__ __align__(16) u16 ldsB[2][128 * 32];   // [buf][n][k]    16KB

  const int t    = threadIdx.x;
  const int lane = t & 31, w = t >> 5;
  const int half = lane >> 4, r16 = lane & 15;
  const int m0 = blockIdx.y * 128, n0 = blockIdx.x * 128;
  const int waveM = w >> 1, waveN = w & 1;

  v8f acc[2][4];
  const v8f vzero = {0.f, 0.f, 0.f, 0.f, 0.f, 0.f, 0.f, 0.f};
#pragma unroll
  for (int i = 0; i < 2; ++i)
#pragma unroll
    for (int j = 0; j < 4; ++j) acc[i][j] = vzero;

  // staging roles (fixed per thread)
  const int rowA = t >> 1, khA = t & 1;     // A: 2 thr/row, 16 f32 each
  const int nB = t & 127, khB = t >> 7;     // B: col n, 16 strided f32

  f32x4 pa[4];
  float pb[16];
  const float* aSrc0 = A + (size_t)(m0 + rowA) * D_DIM + khA * 16;

  // prefetch kk = 0
#pragma unroll
  for (int j = 0; j < 4; ++j) pa[j] = *(const f32x4*)(aSrc0 + 4 * j);
#pragma unroll
  for (int j = 0; j < 16; ++j)
    pb[j] = W[(size_t)(khB * 16 + j) * D_DIM + n0 + nB];

  for (int kk = 0; kk < 32; ++kk) {
    // ---- pack prefetched tile into LDS buf[kk&1] -----------------------
    {
      u32x4* dstA = (u32x4*)&ldsA[kk & 1][rowA * 32 + khA * 16];
      u32x4 p0; p0.x = pk2(pa[0].x, pa[0].y); p0.y = pk2(pa[0].z, pa[0].w);
                p0.z = pk2(pa[1].x, pa[1].y); p0.w = pk2(pa[1].z, pa[1].w);
      u32x4 p1; p1.x = pk2(pa[2].x, pa[2].y); p1.y = pk2(pa[2].z, pa[2].w);
                p1.z = pk2(pa[3].x, pa[3].y); p1.w = pk2(pa[3].z, pa[3].w);
      dstA[0] = p0; dstA[1] = p1;
      u32x4* dstB = (u32x4*)&ldsB[kk & 1][nB * 32 + khB * 16];
      u32x4 q0; q0.x = pk2(pb[0], pb[1]);   q0.y = pk2(pb[2], pb[3]);
                q0.z = pk2(pb[4], pb[5]);   q0.w = pk2(pb[6], pb[7]);
      u32x4 q1; q1.x = pk2(pb[8], pb[9]);   q1.y = pk2(pb[10], pb[11]);
                q1.z = pk2(pb[12], pb[13]); q1.w = pk2(pb[14], pb[15]);
      dstB[0] = q0; dstB[1] = q1;
    }
    __syncthreads();
    // ---- issue global prefetch for kk+1 (overlaps WMMAs below) ---------
    if (kk + 1 < 32) {
      const int k0n = (kk + 1) * 32;
#pragma unroll
      for (int j = 0; j < 4; ++j) pa[j] = *(const f32x4*)(aSrc0 + k0n + 4 * j);
#pragma unroll
      for (int j = 0; j < 16; ++j)
        pb[j] = W[(size_t)(k0n + khB * 16 + j) * D_DIM + n0 + nB];
    }
    // ---- compute from buf[kk&1] ----------------------------------------
    const u16* bufA = ldsA[kk & 1];
    const u16* bufB = ldsB[kk & 1];
    ABFrag af[2], bf[4];
#pragma unroll
    for (int am = 0; am < 2; ++am) {  // A frag: row r, k chunks [8h, 16+8h)
      const int row = waveM * 32 + am * 16 + r16;
      af[am].u[0] = *(const u32x4*)(&bufA[row * 32 + half * 8]);
      af[am].u[1] = *(const u32x4*)(&bufA[row * 32 + 16 + half * 8]);
    }
#pragma unroll
    for (int bn = 0; bn < 4; ++bn) {  // B frag: col n, k chunks [8h, 16+8h)
      const int n = waveN * 64 + bn * 16 + r16;
      bf[bn].u[0] = *(const u32x4*)(&bufB[n * 32 + half * 8]);
      bf[bn].u[1] = *(const u32x4*)(&bufB[n * 32 + 16 + half * 8]);
    }
#pragma unroll
    for (int am = 0; am < 2; ++am)
#pragma unroll
      for (int bn = 0; bn < 4; ++bn)
        acc[am][bn] = wmma_bf16(af[am], bf[bn], acc[am][bn]);
  }

  // epilogue: + bias, cvt bf16, store
#pragma unroll
  for (int am = 0; am < 2; ++am)
#pragma unroll
    for (int bn = 0; bn < 4; ++bn) {
      const int col = n0 + waveN * 64 + bn * 16 + r16;
      const int rowBase = m0 + waveM * 32 + am * 16 + half * 8;
      const float bc = bias[col];
      if (!transpose_v) {
#pragma unroll
        for (int i = 0; i < 8; ++i)
          out[(size_t)(rowBase + i) * D_DIM + col] = f2bf(acc[am][bn][i] + bc);
      } else {
        // Vt[b][d=col][s]; 128-row tiles never straddle a batch boundary
        const int bi = rowBase >> 11, s = rowBase & (S_LEN - 1);
        u32x4 pv;
        pv.x = pk2(acc[am][bn][0] + bc, acc[am][bn][1] + bc);
        pv.y = pk2(acc[am][bn][2] + bc, acc[am][bn][3] + bc);
        pv.z = pk2(acc[am][bn][4] + bc, acc[am][bn][5] + bc);
        pv.w = pk2(acc[am][bn][6] + bc, acc[am][bn][7] + bc);
        *(u32x4*)&out[((size_t)bi * D_DIM + col) * S_LEN + s] = pv;
      }
    }
}

// ---------------------------------------------------------------------------
// Pipelined S-tile loop: s0/s1 += Q[16 x 1024] @ K[256-key block]^T.
// K staged through double-buffered LDS; one barrier per d-step; next d-chunk's
// global loads issued before the frag loads + WMMAs so they stay in flight.
// ---------------------------------------------------------------------------
__device__ __forceinline__ void qk_block(
    const u16* __restrict__ Km, const u16* __restrict__ ldsQ,
    u16* __restrict__ ldsK /* 2 * 256*32 */, size_t keyRowBase,
    int t, int r16, int half, int w, v8f& s0, v8f& s1) {
  const u16* kSrc0 = Km + (keyRowBase + t) * D_DIM;  // this thread's key row
  u32x4 kreg[4];
#pragma unroll
  for (int j = 0; j < 4; ++j) kreg[j] = ((const u32x4*)kSrc0)[j];

  for (int dsi = 0; dsi < 32; ++dsi) {
    u16* buf = ldsK + (dsi & 1) * (256 * 32);
    u32x4* dst = (u32x4*)&buf[t * 32];
#pragma unroll
    for (int j = 0; j < 4; ++j) dst[j] = kreg[j];
    __syncthreads();
    if (dsi + 1 < 32) {
      const u32x4* src = (const u32x4*)(kSrc0 + (dsi + 1) * 32);
#pragma unroll
      for (int j = 0; j < 4; ++j) kreg[j] = src[j];
    }
    ABFrag aQ, b0, b1;
    const u16* qp = ldsQ + r16 * D_DIM + dsi * 32;
    aQ.u[0] = *(const u32x4*)(qp + half * 8);
    aQ.u[1] = *(const u32x4*)(qp + 16 + half * 8);
    const int kr = w * 32 + r16;
    b0.u[0] = *(const u32x4*)(&buf[kr * 32 + half * 8]);
    b0.u[1] = *(const u32x4*)(&buf[kr * 32 + 16 + half * 8]);
    b1.u[0] = *(const u32x4*)(&buf[(kr + 16) * 32 + half * 8]);
    b1.u[1] = *(const u32x4*)(&buf[(kr + 16) * 32 + 16 + half * 8]);
    s0 = wmma_bf16(aQ, b0, s0);
    s1 = wmma_bf16(aQ, b1, s1);
  }
}

// ---------------------------------------------------------------------------
// attn: one WG (8 waves) per (batch, 16-query tile).
//  Pass 1: stats  m,l per row (online softmax over all 2048 keys).
//  Pass 2: recompute S, P = exp(s-m)/l -> LDS, O^T = Vt @ P^T, store fp32.
// ---------------------------------------------------------------------------
__global__ __launch_bounds__(256)
void attn_kernel(const u16* __restrict__ Q, const u16* __restrict__ Km,
                 const u16* __restrict__ Vt, float* __restrict__ O) {
  __shared__ __align__(16) u16 ldsQ[16 * D_DIM];     // 32KB  q-tile [q][d]
  __shared__ __align__(16) u16 ldsK[2][256 * 32];    // 32KB  key blk (dbl buf)
  __shared__ __align__(16) u16 ldsP[16 * 256];       // 8KB   probs [q][key256]
  __shared__ float ldsM[8][16], ldsL[8][16];
  __shared__ float rowMsh[16], rowInvLsh[16];

  const int t    = threadIdx.x;
  const int lane = t & 31, w = t >> 5;
  const int half = lane >> 4, r16 = lane & 15;
  const int q0 = blockIdx.x * 16, b = blockIdx.y;
  const size_t tokBase = (size_t)b * S_LEN;
  const v8f vzero = {0.f, 0.f, 0.f, 0.f, 0.f, 0.f, 0.f, 0.f};

  {  // stage Q tile once (16 x 1024 bf16); first qk_block barrier publishes it
    const int row = t >> 4, ch = t & 15;
    const u32x4* src = (const u32x4*)(Q + (tokBase + q0 + row) * D_DIM + ch * 64);
    u32x4* dst = (u32x4*)&ldsQ[row * D_DIM + ch * 64];
#pragma unroll
    for (int j = 0; j < 8; ++j) dst[j] = src[j];
  }

  // ---------------- pass 1: stats --------------------------------------
  float m_run[8], l_run[8];
#pragma unroll
  for (int i = 0; i < 8; ++i) { m_run[i] = -3.0e38f; l_run[i] = 0.f; }

  for (int kb = 0; kb < 8; ++kb) {
    v8f s0 = vzero, s1 = vzero;
    qk_block(Km, ldsQ, &ldsK[0][0], tokBase + kb * 256, t, r16, half, w, s0, s1);
#pragma unroll
    for (int i = 0; i < 8; ++i) {  // online m,l update (per q row)
      float mx = fmaxf(s0[i], s1[i]);
#pragma unroll
      for (int d = 1; d < 16; d <<= 1) mx = fmaxf(mx, __shfl_xor(mx, d, 32));
      const float mnew = fmaxf(m_run[i], mx);
      float p = __expf(s0[i] - mnew) + __expf(s1[i] - mnew);
#pragma unroll
      for (int d = 1; d < 16; d <<= 1) p += __shfl_xor(p, d, 32);
      l_run[i] = l_run[i] * __expf(m_run[i] - mnew) + p;
      m_run[i] = mnew;
    }
  }
  if (r16 == 0) {  // lanes 0 & 16 publish rows 0-7 / 8-15
#pragma unroll
    for (int i = 0; i < 8; ++i) {
      ldsM[w][half * 8 + i] = m_run[i];
      ldsL[w][half * 8 + i] = l_run[i];
    }
  }
  __syncthreads();
  if (t < 16) {  // combine 8 wave partials per row
    float m = -3.0e38f;
    for (int ww = 0; ww < 8; ++ww) m = fmaxf(m, ldsM[ww][t]);
    float l = 0.f;
    for (int ww = 0; ww < 8; ++ww) l += ldsL[ww][t] * __expf(ldsM[ww][t] - m);
    rowMsh[t] = m;
    rowInvLsh[t] = 1.0f / l;
  }
  __syncthreads();
  float rowM[8], rowInvL[8];
#pragma unroll
  for (int i = 0; i < 8; ++i) {
    rowM[i] = rowMsh[half * 8 + i];
    rowInvL[i] = rowInvLsh[half * 8 + i];
  }

  // ---------------- pass 2: O^T = Vt @ P^T ------------------------------
  v8f o[8];
#pragma unroll
  for (int i = 0; i < 8; ++i) o[i] = vzero;
  const size_t vtBase = (size_t)b * D_DIM * S_LEN;

  for (int kb = 0; kb < 8; ++kb) {
    v8f s0 = vzero, s1 = vzero;
    qk_block(Km, ldsQ, &ldsK[0][0], tokBase + kb * 256, t, r16, half, w, s0, s1);
    {  // P = exp(s - m) / l  -> LDS [q][key] bf16
      const int kc = w * 32 + r16;
#pragma unroll
      for (int i = 0; i < 8; ++i) {
        const int row = half * 8 + i;
        ldsP[row * 256 + kc]      = f2bf(__expf(s0[i] - rowM[i]) * rowInvL[i]);
        ldsP[row * 256 + kc + 16] = f2bf(__expf(s1[i] - rowM[i]) * rowInvL[i]);
      }
    }
    __syncthreads();
    // O^T += Vt(d x key) @ P^T(key x q); A frags straight from global (L2)
    for (int ks = 0; ks < 8; ++ks) {
      ABFrag bP;
      const u16* pp = &ldsP[r16 * 256 + ks * 32];
      bP.u[0] = *(const u32x4*)(pp + half * 8);
      bP.u[1] = *(const u32x4*)(pp + 16 + half * 8);
#pragma unroll
      for (int nt = 0; nt < 8; ++nt) {
        const int drow = w * 128 + nt * 16 + r16;
        const u16* vp = Vt + vtBase + (size_t)drow * S_LEN + kb * 256 + ks * 32;
        ABFrag aV;
        aV.u[0] = *(const u32x4*)(vp + half * 8);
        aV.u[1] = *(const u32x4*)(vp + 16 + half * 8);
        o[nt] = wmma_bf16(aV, bP, o[nt]);
      }
    }
  }

  // epilogue: O^T tile -> O[b][q][d] fp32 (lane holds fixed q, 8 contiguous d)
#pragma unroll
  for (int nt = 0; nt < 8; ++nt) {
    const int qcol = q0 + r16;
    const int dBase = w * 128 + nt * 16 + half * 8;
    float* dst = O + (tokBase + qcol) * D_DIM + dBase;
    f32x4 lo = {o[nt][0], o[nt][1], o[nt][2], o[nt][3]};
    f32x4 hi = {o[nt][4], o[nt][5], o[nt][6], o[nt][7]};
    *(f32x4*)dst = lo;
    *(f32x4*)(dst + 4) = hi;
  }
}

// ---------------------------------------------------------------------------
extern "C" void kernel_launch(void* const* d_in, const int* in_sizes, int n_in,
                              void* d_out, int out_size, void* d_ws, size_t ws_size,
                              hipStream_t stream) {
  const float* x   = (const float*)d_in[0];
  const float* ctx = (const float*)d_in[1];
  const float* wq  = (const float*)d_in[2];
  const float* bq  = (const float*)d_in[3];
  const float* wk  = (const float*)d_in[4];
  const float* bk  = (const float*)d_in[5];
  const float* wv  = (const float*)d_in[6];
  const float* bv  = (const float*)d_in[7];
  // d_in[8] = mask : softmax shift-invariant -> no effect on output.

  const size_t elems = (size_t)NB * S_LEN * D_DIM;  // 8,388,608
  u16* Qb = (u16*)d_ws;
  u16* Kb = Qb + elems;
  u16* Vt = Kb + elems;
  float* out = (float*)d_out;

  dim3 pgrid(D_DIM / 128, (NB * S_LEN) / 128);  // 8 x 64
  proj_gemm_kernel<<<pgrid, 256, 0, stream>>>(x,   wq, bq, Qb, 0);
  proj_gemm_kernel<<<pgrid, 256, 0, stream>>>(ctx, wk, bk, Kb, 0);
  proj_gemm_kernel<<<pgrid, 256, 0, stream>>>(ctx, wv, bv, Vt, 1);

  dim3 agrid(S_LEN / 16, NB);  // 128 x 4
  attn_kernel<<<agrid, 256, 0, stream>>>(Qb, Kb, Vt, out);
}
